// MultiHeadAttention_20684562497876
// MI455X (gfx1250) — matrix operand order repaired and checked by reading, not compiled, not run on hardware
//
#include <hip/hip_runtime.h>
#include <hip/hip_bf16.h>

typedef _Float16 v16h __attribute__((ext_vector_type(16)));
typedef _Float16 v8h  __attribute__((ext_vector_type(8)));
typedef float    v8f  __attribute__((ext_vector_type(8)));

#define BATCH 4
#define SEQ   2048
#define DMODEL 1024
#define HEADS 16
#define DK    64
#define INV_SCALE 0.125f   // 1/sqrt(64)

__device__ __forceinline__ v8f wmma_f16(v16h a, v16h b, v8f c) {
    return __builtin_amdgcn_wmma_f32_16x16x32_f16(
        false, a, false, b, (short)0, c, false, false);
}

__device__ __forceinline__ v16h load_b_f32(const float* bp) {
    float4 b0 = *(const float4*)(bp);
    float4 b1 = *(const float4*)(bp + 4);
    float4 b2 = *(const float4*)(bp + 8);
    float4 b3 = *(const float4*)(bp + 12);
    v16h b;
    b[0]=(_Float16)b0.x; b[1]=(_Float16)b0.y; b[2]=(_Float16)b0.z; b[3]=(_Float16)b0.w;
    b[4]=(_Float16)b1.x; b[5]=(_Float16)b1.y; b[6]=(_Float16)b1.z; b[7]=(_Float16)b1.w;
    b[8]=(_Float16)b2.x; b[9]=(_Float16)b2.y; b[10]=(_Float16)b2.z; b[11]=(_Float16)b2.w;
    b[12]=(_Float16)b3.x; b[13]=(_Float16)b3.y; b[14]=(_Float16)b3.z; b[15]=(_Float16)b3.w;
    return b;
}

// ---------------------------------------------------------------------------
// Projection: Y = X @ W^T + bias,  X:[8192,1024] f32, W:[1024,1024] f32.
// One wave per 16x32 output pair-tile (A fragment reused for 2 WMMAs/chunk).
// TRANS=false: scatter f16 into [B, H, S, DK]   (Q, K)
// TRANS=true : scatter f16 into [B, H, DK, S]   (V, transposed for P@V B-frags)
// ---------------------------------------------------------------------------
template <bool TRANS>
__global__ __launch_bounds__(256)
void proj_kernel(const float* __restrict__ X, const float* __restrict__ W,
                 const float* __restrict__ bias, _Float16* __restrict__ Y)
{
    const int lane = threadIdx.x & 31;
    const int wave = threadIdx.x >> 5;
    const int m0 = blockIdx.x * 16;                   // 512 m-tiles
    const int n0 = (blockIdx.y * 8 + wave) * 32;      // 32 n-pair-tiles
    const int r   = lane & 15;
    const int hi  = lane >> 4;
    const int k0a = hi * 16;   // A-fragment K base for this half-wave
    const int kb  = hi * 16;   // B-fragment K base

    v8f acc0 = {}, acc1 = {};
    for (int kk = 0; kk < DMODEL; kk += 32) {
        // A fragment: row r; halfs 0..15 -> K=k0a+0..15
        const float* ap = X + (size_t)(m0 + r) * DMODEL + kk + k0a;
        float4 a0 = *(const float4*)(ap);
        float4 a1 = *(const float4*)(ap + 4);
        float4 a2 = *(const float4*)(ap + 8);
        float4 a3 = *(const float4*)(ap + 12);
        v16h a;
        a[0]=(_Float16)a0.x; a[1]=(_Float16)a0.y; a[2]=(_Float16)a0.z; a[3]=(_Float16)a0.w;
        a[4]=(_Float16)a1.x; a[5]=(_Float16)a1.y; a[6]=(_Float16)a1.z; a[7]=(_Float16)a1.w;
        a[8]=(_Float16)a2.x; a[9]=(_Float16)a2.y; a[10]=(_Float16)a2.z; a[11]=(_Float16)a2.w;
        a[12]=(_Float16)a3.x; a[13]=(_Float16)a3.y; a[14]=(_Float16)a3.z; a[15]=(_Float16)a3.w;
        // Two B fragments (adjacent n-tiles), 16 contiguous K values each
        v16h b0 = load_b_f32(W + (size_t)(n0 + r) * DMODEL + kk + kb);
        v16h b1 = load_b_f32(W + (size_t)(n0 + 16 + r) * DMODEL + kk + kb);
        acc0 = wmma_f16(a, b0, acc0);
        acc1 = wmma_f16(a, b1, acc1);
    }
    const int bidx  = m0 >> 11;                 // m0 / SEQ
    const int sbase = (m0 & (SEQ - 1)) + hi * 8;
#pragma unroll
    for (int t = 0; t < 2; ++t) {
        const v8f& acc = t ? acc1 : acc0;
        const int n = n0 + t * 16 + r;          // output feature
        const int h = n >> 6, d = n & 63;
        const float bv = bias[n];
        const size_t base = ((size_t)bidx * HEADS + h);
#pragma unroll
        for (int v = 0; v < 8; ++v) {
            const int s = sbase + v;
            const _Float16 val = (_Float16)(acc[v] + bv);
            if (TRANS)
                Y[(base * DK + d) * SEQ + s] = val;      // [B,H,DK,S]
            else
                Y[(base * SEQ + s) * DK + d] = val;      // [B,H,S,DK]
        }
    }
}

// ---------------------------------------------------------------------------
// Fused attention: per (bh, 16-row q strip):
//   scores (WMMA) -> LDS f16 [16][2048] -> softmax -> attn f32 to global
//   -> ctx = P @ V (WMMA, k-split, V pre-transposed) -> ctx f16 [B,S,D]
// ---------------------------------------------------------------------------
__global__ __launch_bounds__(256)
void attn_kernel(const _Float16* __restrict__ Qh, const _Float16* __restrict__ Kh,
                 const _Float16* __restrict__ Vt, float* __restrict__ attn_out,
                 _Float16* __restrict__ ctxh)
{
    __shared__ __align__(16) _Float16 P[16 * SEQ];   // 64 KB
    const int tid  = threadIdx.x;
    const int lane = tid & 31;
    const int wave = tid >> 5;
    const int bh = blockIdx.y;             // b*16 + h
    const int q0 = blockIdx.x * 16;
    const int b  = bh >> 4, h = bh & 15;
    const _Float16* Qbh = Qh + (size_t)bh * SEQ * DK;   // [S, DK]
    const _Float16* Kbh = Kh + (size_t)bh * SEQ * DK;   // [S, DK]
    const _Float16* Vbh = Vt + (size_t)bh * DK * SEQ;   // [DK, S] (transposed)

    const int r   = lane & 15;
    const int hi  = lane >> 4;
    const int k0a = hi * 16;
    const int kb  = hi * 16;

    // Q strip fragments, d-chunks [0,32) and [32,64)
    v16h qa[2];
#pragma unroll
    for (int t = 0; t < 2; ++t) {
        const _Float16* ap = Qbh + (q0 + r) * DK + t * 32 + k0a;
        v8h lo = *(const v8h*)(ap);
        v8h hi8 = *(const v8h*)(ap + 8);
#pragma unroll
        for (int j = 0; j < 8; ++j) { qa[t][j] = lo[j]; qa[t][8 + j] = hi8[j]; }
    }

    // ---- scores: each wave covers 16 column tiles of 16 keys ----
    for (int t = 0; t < 16; ++t) {
        const int col0 = (wave + t * 8) * 16;
        v8f acc = {};
#pragma unroll
        for (int dch = 0; dch < 2; ++dch) {
            const _Float16* bp = Kbh + (col0 + r) * DK + dch * 32 + kb;
            v8h b0 = *(const v8h*)(bp);
            v8h b1 = *(const v8h*)(bp + 8);
            v16h bf;
#pragma unroll
            for (int j = 0; j < 8; ++j) { bf[j] = b0[j]; bf[8 + j] = b1[j]; }
            acc = wmma_f16(qa[dch], bf, acc);
        }
#pragma unroll
        for (int v = 0; v < 8; ++v)
            P[(v + hi * 8) * SEQ + col0 + r] = (_Float16)(acc[v] * INV_SCALE);
    }
    __syncthreads();

    // ---- softmax: 16 lanes per row, width-16 shuffle reductions ----
    {
        const int row = tid >> 4, l16 = tid & 15;
        _Float16* prow = P + row * SEQ;
        float m = -1e30f;
        for (int i = l16; i < SEQ; i += 16) m = fmaxf(m, (float)prow[i]);
#pragma unroll
        for (int off = 8; off; off >>= 1) m = fmaxf(m, __shfl_xor(m, off, 16));
        float sum = 0.f;
        for (int i = l16; i < SEQ; i += 16) {
            float p = __expf((float)prow[i] - m);
            prow[i] = (_Float16)p;
            sum += p;
        }
#pragma unroll
        for (int off = 8; off; off >>= 1) sum += __shfl_xor(sum, off, 16);
        const float inv = 1.f / sum;
        float* arow = attn_out + ((size_t)bh * SEQ + q0 + row) * SEQ;
        for (int i = l16; i < SEQ; i += 16) {
            float p = (float)prow[i] * inv;
            prow[i] = (_Float16)p;      // normalized prob back to LDS for P@V
            arow[i] = p;                // attn output (coalesced 64B/16 lanes)
        }
    }
    __syncthreads();

    // ---- ctx = P[16x2048] @ V[2048x64]: wave -> (d-tile, k-half) ----
    const int ct = wave & 3;           // d tile (0..3)
    const int kh = wave >> 2;          // k half (0..1)
    const int d0 = ct * 16;
    v8f acc = {};
    for (int kc = 0; kc < 32; ++kc) {
        const int kkk = kh * 1024 + kc * 32;
        const _Float16* ap = P + r * SEQ + kkk + k0a;
        v8h lo = *(const v8h*)(ap);
        v8h hi8 = *(const v8h*)(ap + 8);
        v16h af;
#pragma unroll
        for (int j = 0; j < 8; ++j) { af[j] = lo[j]; af[8 + j] = hi8[j]; }
        // B fragment from transposed V: 16 contiguous k per lane -> 2 vector loads
        const _Float16* bp = Vbh + (size_t)(d0 + r) * SEQ + kkk + kb;
        v8h c0 = *(const v8h*)(bp);
        v8h c1 = *(const v8h*)(bp + 8);
        v16h bf;
#pragma unroll
        for (int j = 0; j < 8; ++j) { bf[j] = c0[j]; bf[8 + j] = c1[j]; }
        acc = wmma_f16(af, bf, acc);
    }
    __syncthreads();                   // everyone done reading P
    float* part = (float*)P;           // reuse LDS: [8 waves][256]
#pragma unroll
    for (int v = 0; v < 8; ++v) part[wave * 256 + v * 32 + lane] = acc[v];
    __syncthreads();
    if (wave < 4) {
#pragma unroll
        for (int v = 0; v < 8; ++v) {
            float s2 = part[wave * 256 + v * 32 + lane]
                     + part[(wave + 4) * 256 + v * 32 + lane];
            const int row = v + hi * 8;
            const int d = (wave & 3) * 16 + r;
            ctxh[((size_t)(b * SEQ + q0 + row)) * DMODEL + h * DK + d] = (_Float16)s2;
        }
    }
}

// ---------------------------------------------------------------------------
// Output projection: out = ctx @ w_o^T + b_o,  ctx f16 [8192x1024], out f32.
// One wave per 16x32 pair-tile (A fragment reused for 2 WMMAs/chunk).
// ---------------------------------------------------------------------------
__global__ __launch_bounds__(256)
void out_kernel(const _Float16* __restrict__ Xh, const float* __restrict__ W,
                const float* __restrict__ bias, float* __restrict__ Y)
{
    const int lane = threadIdx.x & 31;
    const int wave = threadIdx.x >> 5;
    const int m0 = blockIdx.x * 16;
    const int n0 = (blockIdx.y * 8 + wave) * 32;
    const int r   = lane & 15;
    const int hi  = lane >> 4;
    const int k0a = hi * 16;
    const int kb  = hi * 16;

    v8f acc0 = {}, acc1 = {};
    for (int kk = 0; kk < DMODEL; kk += 32) {
        const _Float16* ap = Xh + (size_t)(m0 + r) * DMODEL + kk + k0a;
        v8h lo = *(const v8h*)(ap);
        v8h hi8 = *(const v8h*)(ap + 8);
        v16h a;
#pragma unroll
        for (int j = 0; j < 8; ++j) { a[j] = lo[j]; a[8 + j] = hi8[j]; }
        v16h b0 = load_b_f32(W + (size_t)(n0 + r) * DMODEL + kk + kb);
        v16h b1 = load_b_f32(W + (size_t)(n0 + 16 + r) * DMODEL + kk + kb);
        acc0 = wmma_f16(a, b0, acc0);
        acc1 = wmma_f16(a, b1, acc1);
    }
#pragma unroll
    for (int t = 0; t < 2; ++t) {
        const v8f& acc = t ? acc1 : acc0;
        const float bv = bias[n0 + t * 16 + r];
#pragma unroll
        for (int v = 0; v < 8; ++v)
            Y[(size_t)(m0 + hi * 8 + v) * DMODEL + n0 + t * 16 + r] = acc[v] + bv;
    }
}

// ---------------------------------------------------------------------------
extern "C" void kernel_launch(void* const* d_in, const int* in_sizes, int n_in,
                              void* d_out, int out_size, void* d_ws, size_t ws_size,
                              hipStream_t stream)
{
    const float* query = (const float*)d_in[0];
    const float* key_  = (const float*)d_in[1];
    const float* value = (const float*)d_in[2];
    const float* w_q = (const float*)d_in[3];
    const float* b_q = (const float*)d_in[4];
    const float* w_k = (const float*)d_in[5];
    const float* b_k = (const float*)d_in[6];
    const float* w_v = (const float*)d_in[7];
    const float* b_v = (const float*)d_in[8];
    const float* w_o = (const float*)d_in[9];
    const float* b_o = (const float*)d_in[10];

    float* out  = (float*)d_out;                                   // [B,S,D]
    float* attn = out + (size_t)BATCH * SEQ * DMODEL;              // [B,H,S,S]

    const size_t tsz = (size_t)BATCH * SEQ * DMODEL;               // elems per tensor
    _Float16* Qh   = (_Float16*)d_ws;                              // [B,H,S,DK]
    _Float16* Kh   = Qh + tsz;                                     // [B,H,S,DK]
    _Float16* Vth  = Kh + tsz;                                     // [B,H,DK,S]
    _Float16* ctxh = Vth + tsz;                                    // [B,S,D] (67 MB total)

    dim3 gemm_grid(SEQ * BATCH / 16, DMODEL / 32 / 8);             // (512, 4)
    proj_kernel<false><<<gemm_grid, 256, 0, stream>>>(query, w_q, b_q, Qh);
    proj_kernel<false><<<gemm_grid, 256, 0, stream>>>(key_,  w_k, b_k, Kh);
    proj_kernel<true ><<<gemm_grid, 256, 0, stream>>>(value, w_v, b_v, Vth);

    attn_kernel<<<dim3(SEQ / 16, BATCH * HEADS), 256, 0, stream>>>(
        Qh, Kh, Vth, attn, ctxh);

    out_kernel<<<gemm_grid, 256, 0, stream>>>(ctxh, w_o, b_o, out);
}
